// MultiHeadAttention_3135326126386
// MI455X (gfx1250) — matrix operand order repaired and checked by reading, not compiled, not run on hardware
//
#include <hip/hip_runtime.h>

typedef _Float16 v16h __attribute__((ext_vector_type(16)));
typedef _Float16 v8h  __attribute__((ext_vector_type(8)));
typedef _Float16 v4h  __attribute__((ext_vector_type(4)));
typedef float    v8f  __attribute__((ext_vector_type(8)));
typedef unsigned int u32x4 __attribute__((ext_vector_type(4)));
typedef int          i32x8 __attribute__((ext_vector_type(8)));
typedef int          i32x4 __attribute__((ext_vector_type(4)));

#define TSEQ   2048
#define DMODEL 1024
#define NHEAD  16
#define DHEAD  64
#define BATCH  2
#define MTOT   (BATCH * TSEQ)

// ---- LDS weight panel: 64 n-rows x KCHUNK k, f16, bank-swizzle padded ------
// Stream-DWORD p maps to LDS byte 4*p + (p>>6)*8  (8B pad every 64 DWORDs).
// Row stride = 528 B; 528 mod 256 = 16 -> B-fragment lanes (n=0..15) hit 16
// disjoint 16-byte windows across the 64 LDS banks: conflict-free.
#define KCHUNK          256
#define PANEL_DW        (64 * (KCHUNK / 2))        /* 8192 DWORDs            */
#define PANEL_BYTES     (64 * ((KCHUNK * 2) + 16)) /* 33792 B                */

#ifndef __has_builtin
#define __has_builtin(x) 0
#endif
#if __has_builtin(__builtin_amdgcn_tensor_load_to_lds)
#define USE_TDM 1
#else
#define USE_TDM 0
#endif

static __device__ inline v16h pack16(v8h lo, v8h hi) {
    v16h r;
#pragma unroll
    for (int i = 0; i < 8; ++i) { r[i] = lo[i]; r[8 + i] = hi[i]; }
    return r;
}

// B fragment (32x16 f16 KxN) from the padded LDS panel.
// lane n = lane&15 selects column (panel row n_local), kbase=(lane>>4)*16.
static __device__ inline v16h bfrag_lds(const char* ldsBytes, int n_local, int kk) {
    const int p0 = (n_local * KCHUNK + kk) >> 1;        // DWORD stream index
    const char* bp = ldsBytes + p0 * 4 + ((p0 >> 6) * 8);
    v4h x0 = *(const v4h*)(bp);
    v4h x1 = *(const v4h*)(bp + 8);
    v4h x2 = *(const v4h*)(bp + 16);
    v4h x3 = *(const v4h*)(bp + 24);
    v16h r;
#pragma unroll
    for (int i = 0; i < 4; ++i) {
        r[i] = x0[i]; r[4 + i] = x1[i]; r[8 + i] = x2[i]; r[12 + i] = x3[i];
    }
    return r;
}

// Stage a 64-row x KCHUNK-col f16 weight tile into LDS (padded layout).
// TDM path: one tensor_load_to_lds DMA issued by wave 0 with hardware
// padding (pad_interval=64 DW, pad_amount=2 DW) reproducing the swizzle.
static __device__ inline void stage_panel(const _Float16* __restrict__ Wsrc,
                                          int n0, int kc, char* ldsBytes) {
#if USE_TDM
    if ((threadIdx.x >> 5) == 0) {
        const unsigned long long ga =
            (unsigned long long)(uintptr_t)(Wsrc + (size_t)n0 * DMODEL + kc);
        u32x4 g0;
        g0[0] = 1u;                                   // count=1, user mode
        g0[1] = (unsigned int)(uintptr_t)ldsBytes;    // lds_addr
        g0[2] = (unsigned int)ga;                     // global_addr[31:0]
        g0[3] = (unsigned int)((ga >> 32) & 0x01FFFFFFu) | (2u << 30); // type=2
        i32x8 g1;
        g1[0] = (int)((1u << 16)     // data_size = 2B
                    | (1u << 20)     // pad_enable
                    | (5u << 22)     // pad_interval: 64 DWORDs
                    | (1u << 25));   // pad_amount: 2 DWORDs (8B)
        g1[1] = (int)((unsigned)KCHUNK << 16);  // tensor_dim0 = KCHUNK
        g1[2] = (int)(64u << 16);               // tensor_dim1 = 64
        g1[3] = (int)((unsigned)KCHUNK << 16);  // tile_dim0 = KCHUNK
        g1[4] = 64;                             // tile_dim1 = 64, tile_dim2 = 0
        g1[5] = DMODEL;                         // tensor_dim0_stride (elements)
        g1[6] = 0;
        g1[7] = 0;
        i32x4 z4 = {0, 0, 0, 0};                // groups 2/3: unused (2D tensor)
        i32x8 z8 = {0, 0, 0, 0, 0, 0, 0, 0};
        __builtin_amdgcn_tensor_load_to_lds(g0, g1, z4, z4, z8, 0);
#if __has_builtin(__builtin_amdgcn_s_wait_tensorcnt)
        __builtin_amdgcn_s_wait_tensorcnt(0);
#else
        asm volatile("s_wait_tensorcnt 0" ::: "memory");
#endif
    }
#else
    for (int p = threadIdx.x; p < PANEL_DW; p += blockDim.x) {
        const int row = p >> 7;                 // KCHUNK/2 = 128 DW per row
        const int pc  = p & 127;
        const unsigned int d =
            ((const unsigned int*)(Wsrc + (size_t)(n0 + row) * DMODEL + kc))[pc];
        *(unsigned int*)(ldsBytes + p * 4 + ((p >> 6) * 8)) = d;
    }
#endif
    __syncthreads();
}

// ---------------------------------------------------------------------------
// Kernel 0: fp32 -> f16 conversion prepass (x and the four weight matrices).
// ---------------------------------------------------------------------------
__global__ __launch_bounds__(256)
void cvt_kernel(const float* __restrict__ s, _Float16* __restrict__ d, int n) {
    const int i = (blockIdx.x * 256 + threadIdx.x) * 8;
    if (i < n) {
#pragma unroll
        for (int j = 0; j < 8; j += 4) {
            const float4 f = *(const float4*)(s + i + j);
            v4h h;
            h[0] = (_Float16)f.x; h[1] = (_Float16)f.y;
            h[2] = (_Float16)f.z; h[3] = (_Float16)f.w;
            *(v4h*)(d + i + j) = h;
        }
    }
}

// ---------------------------------------------------------------------------
// Kernel 1: fused Q/K/V projection.  y = x @ W^T + b (f16 out).
// Q,K: [b,h,t,d].  V transposed: [b,h,d,t].
// grid (MTOT/128, DMODEL/64, 3), block 256 (8 waves x 16 rows).
// W panel staged per workgroup via TDM, shared by all 8 waves.
// ---------------------------------------------------------------------------
__global__ __launch_bounds__(256)
void qkv_kernel(const _Float16* __restrict__ xh,
                const _Float16* __restrict__ Wqh, const _Float16* __restrict__ Wkh,
                const _Float16* __restrict__ Wvh,
                const float* __restrict__ bq, const float* __restrict__ bk,
                const float* __restrict__ bv,
                _Float16* __restrict__ Q, _Float16* __restrict__ Kh,
                _Float16* __restrict__ Vt) {
    __shared__ __align__(16) char ldsW[PANEL_BYTES];

    const int lane  = threadIdx.x & 31;
    const int wave  = threadIdx.x >> 5;
    const int n15   = lane & 15;
    const int khalf = lane >> 4;
    const int which = blockIdx.z;  // 0=Q 1=K 2=V

    const _Float16* W    = (which == 0) ? Wqh : (which == 1) ? Wkh : Wvh;
    const float*    bias = (which == 0) ? bq  : (which == 1) ? bk  : bv;

    const int m0 = blockIdx.x * 128 + wave * 16;
    const int n0 = blockIdx.y * 64;
    const _Float16* arow = xh + (size_t)(m0 + n15) * DMODEL;

    v8f acc[4] = {};
    for (int c = 0; c < DMODEL / KCHUNK; ++c) {
        const int kc = c * KCHUNK;
        __syncthreads();                    // previous chunk fully consumed
        stage_panel(W, n0, kc, ldsW);       // TDM DMA + barrier
        for (int k0 = 0; k0 < KCHUNK; k0 += 32) {
            v16h a = pack16(*(const v8h*)(arow + kc + k0 + khalf * 16),
                            *(const v8h*)(arow + kc + k0 + khalf * 16 + 8));
#pragma unroll
            for (int t = 0; t < 4; ++t) {
                v16h b = bfrag_lds(ldsW, t * 16 + n15, k0 + khalf * 16);
                acc[t] = __builtin_amdgcn_wmma_f32_16x16x32_f16(
                    false, a, false, b, (short)0, acc[t], false, false);
            }
        }
    }

    const float scale = (which == 0) ? 0.125f : 1.0f;  // 1/sqrt(64) folded into Q
#pragma unroll
    for (int t = 0; t < 4; ++t) {
        const int n  = n0 + t * 16 + n15;
        const float bn = bias[n];
        const int h  = n >> 6;
        const int dd = n & 63;
#pragma unroll
        for (int v = 0; v < 8; ++v) {
            const int rg = m0 + v + 8 * khalf;     // C layout: row = v + 8*(lane>>4)
            const int b  = rg >> 11;
            const int ts = rg & (TSEQ - 1);
            const int bh = b * NHEAD + h;
            const _Float16 val = (_Float16)((acc[t][v] + bn) * scale);
            if (which == 2)
                Vt[((size_t)bh * DHEAD + dd) * TSEQ + ts] = val;
            else if (which == 0)
                Q[((size_t)bh * TSEQ + ts) * DHEAD + dd] = val;
            else
                Kh[((size_t)bh * TSEQ + ts) * DHEAD + dd] = val;
        }
    }
}

// ---------------------------------------------------------------------------
// Kernel 2: causal flash attention.  One wave per (b, h, 16-row q block).
// ---------------------------------------------------------------------------
__global__ __launch_bounds__(32)
void attn_kernel(const _Float16* __restrict__ Q, const _Float16* __restrict__ K,
                 const _Float16* __restrict__ Vt, _Float16* __restrict__ ctx) {
    __shared__ _Float16 ldsP[16 * 32];

    const int lane  = threadIdx.x & 31;
    const int n15   = lane & 15;
    const int khalf = lane >> 4;
    const int q0 = blockIdx.x * 16;
    const int h  = blockIdx.y;
    const int b  = blockIdx.z;
    const int bh = b * NHEAD + h;

    const _Float16* qrow = Q + ((size_t)bh * TSEQ + q0 + n15) * DHEAD;
    const v16h aq0 = pack16(*(const v8h*)(qrow + khalf * 16),
                            *(const v8h*)(qrow + khalf * 16 + 8));
    const v16h aq1 = pack16(*(const v8h*)(qrow + 32 + khalf * 16),
                            *(const v8h*)(qrow + 32 + khalf * 16 + 8));

    v8f o[4] = {};
    float rm[8], rs[8];
#pragma unroll
    for (int v = 0; v < 8; ++v) { rm[v] = -1e30f; rs[v] = 0.0f; }

    const int kend = (q0 + 15) & ~31;
    for (int k0 = 0; k0 <= kend; k0 += 32) {
        v8f s[2] = {};
#pragma unroll
        for (int half = 0; half < 2; ++half) {
            const _Float16* krow =
                K + ((size_t)bh * TSEQ + k0 + half * 16 + n15) * DHEAD + khalf * 16;
            v16h b0 = *(const v16h*)(krow);
            v16h b1 = *(const v16h*)(krow + 32);
            s[half] = __builtin_amdgcn_wmma_f32_16x16x32_f16(false, aq0, false, b0,
                                                             (short)0, s[half], false, false);
            s[half] = __builtin_amdgcn_wmma_f32_16x16x32_f16(false, aq1, false, b1,
                                                             (short)0, s[half], false, false);
        }

        if (k0 + 31 > q0) {
#pragma unroll
            for (int half = 0; half < 2; ++half) {
                const int key = k0 + half * 16 + n15;
#pragma unroll
                for (int v = 0; v < 8; ++v) {
                    const int row = q0 + v + 8 * khalf;
                    if (key > row) s[half][v] = -1e30f;
                }
            }
        }

        float sc[8];
#pragma unroll
        for (int v = 0; v < 8; ++v) {
            float mx = fmaxf(s[0][v], s[1][v]);
#pragma unroll
            for (int m = 1; m <= 8; m <<= 1) mx = fmaxf(mx, __shfl_xor(mx, m, 32));
            const float nm = fmaxf(rm[v], mx);
            sc[v] = __expf(rm[v] - nm);
            rm[v] = nm;
            const float p0 = __expf(s[0][v] - nm);
            const float p1 = __expf(s[1][v] - nm);
            s[0][v] = p0; s[1][v] = p1;
            float ps = p0 + p1;
#pragma unroll
            for (int m = 1; m <= 8; m <<= 1) ps += __shfl_xor(ps, m, 32);
            rs[v] = rs[v] * sc[v] + ps;
        }
#pragma unroll
        for (int t = 0; t < 4; ++t)
#pragma unroll
            for (int v = 0; v < 8; ++v) o[t][v] *= sc[v];

#pragma unroll
        for (int half = 0; half < 2; ++half)
#pragma unroll
            for (int v = 0; v < 8; ++v)
                ldsP[(v + 8 * khalf) * 32 + half * 16 + n15] = (_Float16)s[half][v];
        asm volatile("s_wait_dscnt 0" ::: "memory");  // LDS in-order per wave
        v16h ap = pack16(*(const v8h*)(&ldsP[n15 * 32 + khalf * 16]),
                         *(const v8h*)(&ldsP[n15 * 32 + khalf * 16 + 8]));

#pragma unroll
        for (int t = 0; t < 4; ++t) {
            const _Float16* vrow =
                Vt + ((size_t)bh * DHEAD + t * 16 + n15) * TSEQ + k0 + khalf * 16;
            v16h bv = *(const v16h*)(vrow);
            o[t] = __builtin_amdgcn_wmma_f32_16x16x32_f16(false, ap, false, bv,
                                                          (short)0, o[t], false, false);
        }
    }

#pragma unroll
    for (int t = 0; t < 4; ++t)
#pragma unroll
        for (int v = 0; v < 8; ++v) {
            const int row = q0 + v + 8 * khalf;
            const float val = o[t][v] / rs[v];
            ctx[((size_t)b * TSEQ + row) * DMODEL + h * DHEAD + t * 16 + n15] = (_Float16)val;
        }
}

// ---------------------------------------------------------------------------
// Kernel 3: output projection.  out = ctx @ Wo^T + bo (fp32 out).
// grid (MTOT/128, DMODEL/64), block 256; Wo panel via TDM.
// ---------------------------------------------------------------------------
__global__ __launch_bounds__(256)
void oproj_kernel(const _Float16* __restrict__ ctx, const _Float16* __restrict__ Woh,
                  const float* __restrict__ bo, float* __restrict__ out) {
    __shared__ __align__(16) char ldsW[PANEL_BYTES];

    const int lane  = threadIdx.x & 31;
    const int wave  = threadIdx.x >> 5;
    const int n15   = lane & 15;
    const int khalf = lane >> 4;
    const int m0 = blockIdx.x * 128 + wave * 16;
    const int n0 = blockIdx.y * 64;
    const _Float16* arow = ctx + (size_t)(m0 + n15) * DMODEL;

    v8f acc[4] = {};
    for (int c = 0; c < DMODEL / KCHUNK; ++c) {
        const int kc = c * KCHUNK;
        __syncthreads();
        stage_panel(Woh, n0, kc, ldsW);
        for (int k0 = 0; k0 < KCHUNK; k0 += 32) {
            v16h a = pack16(*(const v8h*)(arow + kc + k0 + khalf * 16),
                            *(const v8h*)(arow + kc + k0 + khalf * 16 + 8));
#pragma unroll
            for (int t = 0; t < 4; ++t) {
                v16h b = bfrag_lds(ldsW, t * 16 + n15, k0 + khalf * 16);
                acc[t] = __builtin_amdgcn_wmma_f32_16x16x32_f16(
                    false, a, false, b, (short)0, acc[t], false, false);
            }
        }
    }
#pragma unroll
    for (int t = 0; t < 4; ++t) {
        const int n  = n0 + t * 16 + n15;
        const float bn = bo[n];
#pragma unroll
        for (int v = 0; v < 8; ++v) {
            const int rg = m0 + v + 8 * khalf;
            out[(size_t)rg * DMODEL + n] = acc[t][v] + bn;
        }
    }
}

// ---------------------------------------------------------------------------
extern "C" void kernel_launch(void* const* d_in, const int* in_sizes, int n_in,
                              void* d_out, int out_size, void* d_ws, size_t ws_size,
                              hipStream_t stream) {
    const float* x  = (const float*)d_in[0];
    const float* Wq = (const float*)d_in[1];
    const float* bq = (const float*)d_in[2];
    const float* Wk = (const float*)d_in[3];
    const float* bk = (const float*)d_in[4];
    const float* Wv = (const float*)d_in[5];
    const float* bv = (const float*)d_in[6];
    const float* Wo = (const float*)d_in[7];
    const float* bo = (const float*)d_in[8];
    float* out = (float*)d_out;

    char* ws = (char*)d_ws;
    const size_t xBytes = (size_t)MTOT * DMODEL * sizeof(_Float16);            // 8 MB
    const size_t wBytes = (size_t)DMODEL * DMODEL * sizeof(_Float16);          // 2 MB
    const size_t hBytes = (size_t)BATCH * NHEAD * TSEQ * DHEAD * sizeof(_Float16); // 8 MB

    _Float16* xh  = (_Float16*)(ws);
    _Float16* Wqh = (_Float16*)(ws + xBytes);
    _Float16* Wkh = (_Float16*)(ws + xBytes + wBytes);
    _Float16* Wvh = (_Float16*)(ws + xBytes + 2 * wBytes);
    _Float16* Woh = (_Float16*)(ws + xBytes + 3 * wBytes);
    _Float16* Q   = (_Float16*)(ws + xBytes + 4 * wBytes);
    _Float16* Kh  = (_Float16*)(ws + xBytes + 4 * wBytes + hBytes);
    _Float16* Vt  = (_Float16*)(ws + xBytes + 4 * wBytes + 2 * hBytes);
    _Float16* ctx = (_Float16*)(ws + xBytes + 4 * wBytes + 3 * hBytes);

    const int nx = MTOT * DMODEL;     // 4,194,304
    const int nw = DMODEL * DMODEL;   // 1,048,576
    cvt_kernel<<<nx / 2048, 256, 0, stream>>>(x,  xh,  nx);
    cvt_kernel<<<nw / 2048, 256, 0, stream>>>(Wq, Wqh, nw);
    cvt_kernel<<<nw / 2048, 256, 0, stream>>>(Wk, Wkh, nw);
    cvt_kernel<<<nw / 2048, 256, 0, stream>>>(Wv, Wvh, nw);
    cvt_kernel<<<nw / 2048, 256, 0, stream>>>(Wo, Woh, nw);

    dim3 g1(MTOT / 128, DMODEL / 64, 3);
    qkv_kernel<<<g1, 256, 0, stream>>>(xh, Wqh, Wkh, Wvh, bq, bk, bv, Q, Kh, Vt);

    dim3 g2(TSEQ / 16, NHEAD, BATCH);
    attn_kernel<<<g2, 32, 0, stream>>>(Q, Kh, Vt, ctx);

    dim3 g3(MTOT / 128, DMODEL / 64);
    oproj_kernel<<<g3, 256, 0, stream>>>(ctx, Woh, bo, out);
}
